// AttentionLayer_26637387170242
// MI455X (gfx1250) — matrix operand order repaired and checked by reading, not compiled, not run on hardware
//
#include <hip/hip_runtime.h>

#define NB     2
#define LSEQ   2048
#define DMODEL 512
#define NH     8
#define HDIM   64
#define SROW   2052   // padded LDS row stride (floats) for 2048-wide rows
#define VROW   65     // staged V row stride (floats): 64 + 1 pad (TDM pad_enable)

typedef __attribute__((ext_vector_type(16))) _Float16     v16h;
typedef __attribute__((ext_vector_type(8)))  float        v8f;
typedef __attribute__((ext_vector_type(4)))  unsigned int u32x4;
typedef __attribute__((ext_vector_type(8)))  int          i32x8;
typedef __attribute__((ext_vector_type(4)))  int          i32x4;

// A-matrix fragment (16x32 f16): lane-half hl holds contiguous K = hl*16 .. hl*16+15
__device__ __forceinline__ v16h fragA_f32(const float* __restrict__ p, int hl) {
    const float4 a = *(const float4*)(p + hl * 16);
    const float4 b = *(const float4*)(p + hl * 16 + 4);
    const float4 c = *(const float4*)(p + hl * 16 + 8);
    const float4 d = *(const float4*)(p + hl * 16 + 12);
    v16h f;
    f[0]  = (_Float16)a.x; f[1]  = (_Float16)a.y; f[2]  = (_Float16)a.z; f[3]  = (_Float16)a.w;
    f[4]  = (_Float16)b.x; f[5]  = (_Float16)b.y; f[6]  = (_Float16)b.z; f[7]  = (_Float16)b.w;
    f[8]  = (_Float16)c.x; f[9]  = (_Float16)c.y; f[10] = (_Float16)c.z; f[11] = (_Float16)c.w;
    f[12] = (_Float16)d.x; f[13] = (_Float16)d.y; f[14] = (_Float16)d.z; f[15] = (_Float16)d.w;
    return f;
}

// B-matrix fragment (32x16 f16): lane-half hl holds contiguous K = hl*16 .. hl*16+15
__device__ __forceinline__ v16h fragB_f32(const float* __restrict__ p, int hl) {
    const float4 a = *(const float4*)(p + hl * 16);
    const float4 b = *(const float4*)(p + hl * 16 + 4);
    const float4 c = *(const float4*)(p + hl * 16 + 8);
    const float4 d = *(const float4*)(p + hl * 16 + 12);
    v16h f;
    f[0]  = (_Float16)a.x; f[1]  = (_Float16)a.y; f[2]  = (_Float16)a.z; f[3]  = (_Float16)a.w;
    f[4]  = (_Float16)b.x; f[5]  = (_Float16)b.y; f[6]  = (_Float16)b.z; f[7]  = (_Float16)b.w;
    f[8]  = (_Float16)c.x; f[9]  = (_Float16)c.y; f[10] = (_Float16)c.z; f[11] = (_Float16)c.w;
    f[12] = (_Float16)d.x; f[13] = (_Float16)d.y; f[14] = (_Float16)d.z; f[15] = (_Float16)d.w;
    return f;
}

// Issue one TDM load: 64 rows x 64 f32 cols of x (row stride DMODEL) -> LDS,
// LDS rows padded to 65 dwords via pad_enable (interval=64 dwords, amount=1 dword).
__device__ __forceinline__ void tdm_stage_v(const float* gsrc, unsigned lds_off_bytes) {
    const unsigned long long ga = (unsigned long long)(const void*)gsrc;
    u32x4 g0;
    g0[0] = 1u;                                            // count=1, user descriptor
    g0[1] = lds_off_bytes;                                 // lds_addr (bytes)
    g0[2] = (unsigned)(ga & 0xFFFFFFFFu);                  // global_addr[31:0]
    g0[3] = (unsigned)((ga >> 32) & 0x01FFFFFFu)           // global_addr[56:32]
          | 0x80000000u;                                   // type=2 ("image")
    i32x8 g1;
    g1[0] = 0x01520000;            // data_size=4B | pad_enable | pad_interval=64dw | pad=1dw
    g1[1] = (int)(512u  << 16);    // tensor_dim0 = DMODEL (data_size units)
    g1[2] = (int)(2048u << 16);    // tensor_dim1 = LSEQ
    g1[3] = (int)(64u   << 16);    // tile_dim0  = 64 cols
    g1[4] = 64;                    // tile_dim1  = 64 rows (tile_dim2 = 0)
    g1[5] = 512;                   // tensor_dim0_stride = DMODEL
    g1[6] = 0;
    g1[7] = 0;
    const i32x4 z4 = {0, 0, 0, 0};
    const i32x8 z8 = {0, 0, 0, 0, 0, 0, 0, 0};
    // clang-23 / amdgpu-toolchain 6-arg form:
    // (g0, g1, g2, g3, extra_group, cpol)
    __builtin_amdgcn_tensor_load_to_lds(g0, g1, z4, z4, z8, 0);
}

__global__ __launch_bounds__(128)
void attn_wmma_kernel(const float* __restrict__ x,
                      float* __restrict__ out,
                      float* __restrict__ attn) {
    extern __shared__ float smem[];
    float* sS   = smem;                       // 16 x SROW  : scores / probabilities
    float* sAcc = smem + 16 * SROW;           // 16 x SROW  : head-mean accumulator
    float* sRed = smem + 32 * SROW;           // 16 x 8     : row-reduction scratch
    float* sV0  = smem + 32 * SROW + 128;     // 64 x VROW  : V stage buffer 0
    float* sV1  = sV0 + 64 * VROW;            // 64 x VROW  : V stage buffer 1

    const int tid  = threadIdx.x;
    const int lane = tid & 31;
    const int wv   = tid >> 5;        // wave 0..3
    const int hl   = lane >> 4;       // lane half
    const int mn   = lane & 15;       // M (A/C row) or N (B/C col) index

    const int nb = blockIdx.x >> 7;             // batch
    const int i0 = (blockIdx.x & 127) << 4;     // query-row tile base
    const float scale = 0.125f;                 // 1/sqrt(64)

    const float* xb = x + (size_t)nb * LSEQ * DMODEL;
    float* sV[2] = { sV0, sV1 };
    const unsigned sVoff[2] = { (unsigned)(size_t)sV0, (unsigned)(size_t)sV1 };

    for (int h = 0; h < NH; ++h) {
        const int hc = h * HDIM;

        // ---------- Phase 1: S = Q K^T (16 x 2048), f16 WMMA, f32 accum ----------
        const float* qrow = xb + (size_t)(i0 + mn) * DMODEL + hc;
        const v16h aQ0 = fragA_f32(qrow, hl);
        const v16h aQ1 = fragA_f32(qrow + 32, hl);

        const int st0 = wv * 32, st1 = st0 + 32;
        for (int st = st0; st < st1; ++st) {
            const int s0 = st << 4;
            const float* krow = xb + (size_t)(s0 + mn) * DMODEL + hc;
            if (s0 + 16 < LSEQ) __builtin_prefetch(krow + 16 * DMODEL, 0, 1);
            const v16h bK0 = fragB_f32(krow, hl);
            const v16h bK1 = fragB_f32(krow + 32, hl);
            v8f c = {};
            c = __builtin_amdgcn_wmma_f32_16x16x32_f16(false, aQ0, false, bK0,
                                                       (short)0, c, false, false);
            c = __builtin_amdgcn_wmma_f32_16x16x32_f16(false, aQ1, false, bK1,
                                                       (short)0, c, false, false);
            float* dst = sS + (size_t)(8 * hl) * SROW + s0 + mn;   // M = j + 8*hl, N = mn
#pragma unroll
            for (int j = 0; j < 8; ++j) dst[(size_t)j * SROW] = c[j];
        }
        __syncthreads();

        // ---------- Phase 2: softmax rows + fold 1/H mean into sAcc ----------
        {
            const int r = tid >> 3, sub = tid & 7;     // 8 threads per row
            float* row = sS + (size_t)r * SROW;
            float lm = -3.4e38f;
            for (int cix = sub; cix < LSEQ; cix += 8) lm = fmaxf(lm, row[cix]);
            sRed[r * 8 + sub] = lm;
            __syncthreads();
            float rm = -3.4e38f;
#pragma unroll
            for (int j = 0; j < 8; ++j) rm = fmaxf(rm, sRed[r * 8 + j]);
            __syncthreads();
            float ls = 0.f;
            for (int cix = sub; cix < LSEQ; cix += 8) {
                const float e = __expf((row[cix] - rm) * scale);
                row[cix] = e;
                ls += e;
            }
            sRed[r * 8 + sub] = ls;
            __syncthreads();
            float rs = 0.f;
#pragma unroll
            for (int j = 0; j < 8; ++j) rs += sRed[r * 8 + j];
            const float inv = 1.0f / rs;
            float* arow = sAcc + (size_t)r * SROW;
            for (int cix = sub; cix < LSEQ; cix += 8) {
                const float p = row[cix] * inv;
                row[cix] = p;
                const float a = p * 0.125f;            // mean over 8 heads
                arow[cix] = (h == 0) ? a : (arow[cix] + a);
            }
        }
        __syncthreads();

        // ---------- Phase 3: O = P V via WMMA, V staged by TDM (double-buffered) ----------
        {
            const int d0 = wv << 4;
            v8f o = {};
            const int NSTG = LSEQ / 64;                      // 32 stages of 64 keys
            if (wv == 0)                                     // prime the pipeline
                tdm_stage_v(xb + hc, sVoff[0]);
            for (int stg = 0; stg < NSTG; ++stg) {
                const int ks = stg * 64;
                if (wv == 0) {
                    if (stg + 1 < NSTG) {
                        tdm_stage_v(xb + (size_t)(ks + 64) * DMODEL + hc,
                                    sVoff[(stg + 1) & 1]);
                        __builtin_amdgcn_s_wait_tensorcnt(1);  // stage stg done (in-order)
                    } else {
                        __builtin_amdgcn_s_wait_tensorcnt(0);
                    }
                }
                __syncthreads();                             // publish staged V tile
                const float* vb   = sV[stg & 1];
                const float* prow = sS + (size_t)mn * SROW + ks;
#pragma unroll
                for (int c2 = 0; c2 < 2; ++c2) {
                    const v16h aP = fragA_f32(prow + c2 * 32, hl);
                    v16h bV;
                    const float* vp = vb + (size_t)(c2 * 32 + hl * 16) * VROW + d0 + mn;
#pragma unroll
                    for (int e = 0; e < 16; ++e) bV[e] = (_Float16)vp[(size_t)e * VROW];
                    o = __builtin_amdgcn_wmma_f32_16x16x32_f16(false, aP, false, bV,
                                                               (short)0, o, false, false);
                }
                __syncthreads();                             // buffer reuse guard
            }
            float* orow = out + ((size_t)nb * LSEQ + i0 + 8 * hl) * DMODEL + hc + d0 + mn;
#pragma unroll
            for (int j = 0; j < 8; ++j) orow[(size_t)j * DMODEL] = o[j];
        }
        __syncthreads();
    }

    // ---------- Phase 4: flush head-averaged attention strip ----------
    float* ab = attn + ((size_t)nb * LSEQ + i0) * LSEQ;
    for (int idx = tid; idx < 16 * LSEQ; idx += 128) {
        const int r   = idx >> 11;
        const int cix = idx & (LSEQ - 1);
        ab[(size_t)r * LSEQ + cix] = sAcc[(size_t)r * SROW + cix];
    }
}

extern "C" void kernel_launch(void* const* d_in, const int* in_sizes, int n_in,
                              void* d_out, int out_size, void* d_ws, size_t ws_size,
                              hipStream_t stream) {
    (void)in_sizes; (void)n_in; (void)out_size; (void)d_ws; (void)ws_size;
    const float* x = (const float*)d_in[0];
    float* out  = (float*)d_out;
    float* attn = out + (size_t)NB * LSEQ * DMODEL;   // outputs concatenated: (out, attn)

    // 2x(16xSROW) score/mean + reduction scratch + 2x(64xVROW) V stage: ~290 KB < 320 KB
    const size_t smemBytes =
        (size_t)(32 * SROW + 16 * 8 + 2 * 64 * VROW) * sizeof(float);
    (void)hipFuncSetAttribute(reinterpret_cast<const void*>(attn_wmma_kernel),
                              hipFuncAttributeMaxDynamicSharedMemorySize,
                              (int)smemBytes);

    const dim3 grid(NB * (LSEQ / 16));   // 256 workgroups, one 16-row query tile each
    const dim3 block(128);               // 4 waves (wave32)
    attn_wmma_kernel<<<grid, block, smemBytes, stream>>>(x, out, attn);
}
